// EGCL_19198503813801
// MI455X (gfx1250) — matrix operand order repaired and checked by reading, not compiled, not run on hardware
//
#include <hip/hip_runtime.h>
#include <math.h>

#define NNODES 20000
#define NEDGES 640000
#define FEATD  128
#define HIDD   128
#define KPAD   136              // padded row length (halves) to dodge bank conflicts
#define NTILES_E (NEDGES/16)
#define NTILES_N (NNODES/16)

typedef _Float16 h16 __attribute__((ext_vector_type(16)));
typedef _Float16 h8v __attribute__((ext_vector_type(8)));
typedef float    f8  __attribute__((ext_vector_type(8)));
typedef float    f4v __attribute__((ext_vector_type(4)));

#define DSFENCE() asm volatile("s_wait_dscnt 0" ::: "memory")

__device__ __forceinline__ f8 wmma_f16(h16 a, h16 b, f8 c) {
  return __builtin_amdgcn_wmma_f32_16x16x32_f16(false, a, false, b, (short)0, c, false, false);
}

// B fragment: lane L holds N=L%16, K = 32c + 8*(L/16) + {0..7, 16..23}, from LDS Wt[n][k] (f16)
__device__ __forceinline__ h16 ldsB(const _Float16* W, int nrow, int c, int g) {
  const _Float16* p = W + nrow * KPAD + c * 32 + g * 8;
  h8v lo = *(const h8v*)p;
  h8v hi = *(const h8v*)(p + 16);
  h16 r;
#pragma unroll
  for (int i = 0; i < 8; ++i) { r[i] = lo[i]; r[i + 8] = hi[i]; }
  return r;
}

// A fragment from per-wave LDS staging [16][KPAD] f16 (row = edge/node in tile)
// lane L<16: K = 32c+{0..7, 16..23};  lane L>=16: K = 32c+{8..15, 24..31}
__device__ __forceinline__ h16 ldsA(const _Float16* S, int lm, int g, int c) {
  const _Float16* p = S + lm * KPAD + c * 32 + g * 8;
  h8v lo = *(const h8v*)p;
  h8v hi = *(const h8v*)(p + 16);
  h16 r;
#pragma unroll
  for (int i = 0; i < 8; ++i) { r[i] = lo[i]; r[i + 8] = hi[i]; }
  return r;
}

// A fragment built directly from a global f32 row (first-layer inputs)
__device__ __forceinline__ h16 gfragA(const float* row, int c, int g) {
  const float* p = row + c * 32 + g * 8;
  f4v a0 = *(const f4v*)(p);
  f4v a1 = *(const f4v*)(p + 4);
  f4v a2 = *(const f4v*)(p + 16);
  f4v a3 = *(const f4v*)(p + 20);
  h16 r;
#pragma unroll
  for (int i = 0; i < 4; ++i) {
    r[i]      = (_Float16)a0[i];
    r[4 + i]  = (_Float16)a1[i];
    r[8 + i]  = (_Float16)a2[i];
    r[12 + i] = (_Float16)a3[i];
  }
  return r;
}

__device__ __forceinline__ float silu_f(float x) { return x / (1.f + __expf(-x)); }

// transpose-store a 128x128 f32 weight (row-major [k][n]) into LDS f16 Wt[n][KPAD]
__device__ __forceinline__ void load_wt(_Float16* dst, const float* src) {
  for (int i = threadIdx.x; i < 128 * 128; i += blockDim.x) {
    int k = i >> 7, n = i & 127;
    dst[n * KPAD + k] = (_Float16)src[i];
  }
}

// ---------------- init: seed d_out positions, zero m_i accumulator ----------------
__global__ __launch_bounds__(256) void egcl_init(const float* __restrict__ pos,
                                                 float* __restrict__ pos_out,
                                                 float* __restrict__ mi) {
  int i = blockIdx.x * blockDim.x + threadIdx.x;
  if (i < NNODES * 3) pos_out[i] = pos[i];
  for (int k = i; k < NNODES * HIDD; k += gridDim.x * blockDim.x) mi[k] = 0.f;
}

// ---------------- edge kernel: phi_e, gate, phi_x, scatters ----------------
__global__ __launch_bounds__(256) void egcl_edge(
    const float* __restrict__ pos, const float* __restrict__ nf,
    const float* __restrict__ pe_w0, const float* __restrict__ pe_b0,
    const float* __restrict__ pe_w1, const float* __restrict__ pe_b1,
    const float* __restrict__ px_w0, const float* __restrict__ px_b0,
    const float* __restrict__ px_w1, const float* __restrict__ px_b1,
    const float* __restrict__ px_ow, const float* __restrict__ px_ob,
    const float* __restrict__ e_w, const float* __restrict__ e_b,
    const int* __restrict__ senders, const int* __restrict__ receivers,
    float* __restrict__ mi, float* __restrict__ pos_out) {
  extern __shared__ _Float16 sm[];
  _Float16* W0s = sm;
  _Float16* W0r = sm + 1 * 128 * KPAD;
  _Float16* W1  = sm + 2 * 128 * KPAD;
  _Float16* Wx0 = sm + 3 * 128 * KPAD;
  _Float16* Wx1 = sm + 4 * 128 * KPAD;
  load_wt(W0s, pe_w0);               // edge_in senders block   (rows 0..127)
  load_wt(W0r, pe_w0 + 128 * 128);   // edge_in receivers block (rows 128..255)
  load_wt(W1, pe_w1);
  load_wt(Wx0, px_w0);
  load_wt(Wx1, px_w1);
  __syncthreads();

  const int lane = threadIdx.x & 31;
  const int wave = threadIdx.x >> 5;
  const int g = lane >> 4;
  const int lm = lane & 15;
  _Float16* SA = sm + 5 * 128 * KPAD + wave * 16 * KPAD;

  const int wid = blockIdx.x * 8 + wave;
  const int nw = gridDim.x * 8;
  const float inv_avg = 1.0f / (float)(NNODES - 1);
  const float inv_sqrt_avg = rsqrtf((float)(NNODES - 1));

  for (int t = wid; t < NTILES_E; t += nw) {
    const int e = t * 16 + lm;
    const int sidx = senders[e];
    const int ridx = receivers[e];
    const float vx = pos[ridx * 3 + 0] - pos[sidx * 3 + 0];
    const float vy = pos[ridx * 3 + 1] - pos[sidx * 3 + 1];
    const float vz = pos[ridx * 3 + 2] - pos[sidx * 3 + 2];
    const float n2 = vx * vx + vy * vy + vz * vz;
    const float len = (n2 > 0.f) ? sqrtf(n2) : 0.f;

    // A fragments: sender & receiver feature rows (gathered from global/L2)
    h16 As[4], Ar[4];
    {
      const float* srow = nf + (size_t)sidx * HIDD;
      const float* rrow = nf + (size_t)ridx * HIDD;
#pragma unroll
      for (int c = 0; c < 4; ++c) { As[c] = gfragA(srow, c, g); Ar[c] = gfragA(rrow, c, g); }
    }

    // ---- phi_e layer 0:  silu(Fs@W0s + Fr@W0r + len*w0[256] + b0) ----
#pragma unroll
    for (int nt = 0; nt < 8; ++nt) {
      const int n = nt * 16 + lm;
      const float b = pe_b0[n];
      const float wl = pe_w0[256 * 128 + n];
      f8 acc;
#pragma unroll
      for (int j = 0; j < 8; ++j) { float lj = __shfl(len, j + 8 * g); acc[j] = b + lj * wl; }
#pragma unroll
      for (int c = 0; c < 4; ++c) acc = wmma_f16(As[c], ldsB(W0s, n, c, g), acc);
#pragma unroll
      for (int c = 0; c < 4; ++c) acc = wmma_f16(Ar[c], ldsB(W0r, n, c, g), acc);
#pragma unroll
      for (int j = 0; j < 8; ++j) SA[(j + 8 * g) * KPAD + n] = (_Float16)silu_f(acc[j]);
    }

    DSFENCE();
    h16 Am[4];
#pragma unroll
    for (int c = 0; c < 4; ++c) Am[c] = ldsA(SA, lm, g, c);
    DSFENCE();

    // ---- phi_e layer 1 -> m_ij (kept in regs AND re-staged as next A) ----
    f8 macc[8];
#pragma unroll
    for (int nt = 0; nt < 8; ++nt) {
      const int n = nt * 16 + lm;
      const float b = pe_b1[n];
      f8 acc;
#pragma unroll
      for (int j = 0; j < 8; ++j) acc[j] = b;
#pragma unroll
      for (int c = 0; c < 4; ++c) acc = wmma_f16(Am[c], ldsB(W1, n, c, g), acc);
#pragma unroll
      for (int j = 0; j < 8; ++j) {
        float x = silu_f(acc[j]);
        macc[nt][j] = x;
        SA[(j + 8 * g) * KPAD + n] = (_Float16)x;
      }
    }

    // ---- gate e = sigmoid(m_ij @ e_w + e_b): butterfly dot over the 16-lane halves
    float ps[8];
#pragma unroll
    for (int j = 0; j < 8; ++j) ps[j] = 0.f;
#pragma unroll
    for (int nt = 0; nt < 8; ++nt) {
      float w = e_w[nt * 16 + lm];
#pragma unroll
      for (int j = 0; j < 8; ++j) ps[j] += macc[nt][j] * w;
    }
#pragma unroll
    for (int m = 1; m < 16; m <<= 1)
#pragma unroll
      for (int j = 0; j < 8; ++j) ps[j] += __shfl_xor(ps[j], m);
    const float ebv = e_b[0];
    float eg[8];
#pragma unroll
    for (int j = 0; j < 8; ++j) eg[j] = 1.f / (1.f + __expf(-(ps[j] + ebv)));

    int rj[8];
#pragma unroll
    for (int j = 0; j < 8; ++j) rj[j] = __shfl(ridx, j + 8 * g);

    // ---- scatter gated messages, pre-scaled by 1/sqrt(N-1) ----
#pragma unroll
    for (int nt = 0; nt < 8; ++nt)
#pragma unroll
      for (int j = 0; j < 8; ++j) {
        float v = macc[nt][j] * eg[j] * inv_sqrt_avg;
        unsafeAtomicAdd(&mi[(size_t)rj[j] * HIDD + nt * 16 + lm], v);
      }

    // ---- phi_x layer 0 (input = m_ij from staging) ----
    DSFENCE();
#pragma unroll
    for (int c = 0; c < 4; ++c) Am[c] = ldsA(SA, lm, g, c);
    DSFENCE();
#pragma unroll
    for (int nt = 0; nt < 8; ++nt) {
      const int n = nt * 16 + lm;
      const float b = px_b0[n];
      f8 acc;
#pragma unroll
      for (int j = 0; j < 8; ++j) acc[j] = b;
#pragma unroll
      for (int c = 0; c < 4; ++c) acc = wmma_f16(Am[c], ldsB(Wx0, n, c, g), acc);
#pragma unroll
      for (int j = 0; j < 8; ++j) SA[(j + 8 * g) * KPAD + n] = (_Float16)silu_f(acc[j]);
    }

    // ---- phi_x layer 1 + Dense(1) head, fused into a dot ----
    DSFENCE();
#pragma unroll
    for (int c = 0; c < 4; ++c) Am[c] = ldsA(SA, lm, g, c);
    DSFENCE();
    float pd[8];
#pragma unroll
    for (int j = 0; j < 8; ++j) pd[j] = 0.f;
#pragma unroll
    for (int nt = 0; nt < 8; ++nt) {
      const int n = nt * 16 + lm;
      const float b = px_b1[n];
      f8 acc;
#pragma unroll
      for (int j = 0; j < 8; ++j) acc[j] = b;
#pragma unroll
      for (int c = 0; c < 4; ++c) acc = wmma_f16(Am[c], ldsB(Wx1, n, c, g), acc);
      const float w = px_ow[n];
#pragma unroll
      for (int j = 0; j < 8; ++j) pd[j] += silu_f(acc[j]) * w;
    }
#pragma unroll
    for (int m = 1; m < 16; m <<= 1)
#pragma unroll
      for (int j = 0; j < 8; ++j) pd[j] += __shfl_xor(pd[j], m);
    const float pob = px_ob[0];

    // ---- shift scatter: shifts = phi_x_out * vec / (1+len) / (N-1) ----
#pragma unroll
    for (int j = 0; j < 8; ++j) {
      const int M = j + 8 * g;
      float vxj = __shfl(vx, M), vyj = __shfl(vy, M), vzj = __shfl(vz, M);
      float lj = __shfl(len, M);
      if (lm == j) {
        float f = (pd[j] + pob) * inv_avg / (1.f + lj);
        unsafeAtomicAdd(&pos_out[(size_t)rj[j] * 3 + 0], f * vxj);
        unsafeAtomicAdd(&pos_out[(size_t)rj[j] * 3 + 1], f * vyj);
        unsafeAtomicAdd(&pos_out[(size_t)rj[j] * 3 + 2], f * vzj);
      }
    }
  }
}

// ---------------- node kernel: phi_h + residual ----------------
__global__ __launch_bounds__(256) void egcl_node(
    const float* __restrict__ nf, const float* __restrict__ mi,
    const float* __restrict__ ph_w0, const float* __restrict__ ph_b0,
    const float* __restrict__ ph_w1, const float* __restrict__ ph_b1,
    const float* __restrict__ ph_w2, const float* __restrict__ ph_b2,
    float* __restrict__ feat_out) {
  extern __shared__ _Float16 sm[];
  _Float16* W0a = sm;                    // m_i block   (rows 0..127)
  _Float16* W0b = sm + 1 * 128 * KPAD;   // features block (rows 128..255)
  _Float16* W1  = sm + 2 * 128 * KPAD;
  _Float16* W2  = sm + 3 * 128 * KPAD;
  load_wt(W0a, ph_w0);
  load_wt(W0b, ph_w0 + 128 * 128);
  load_wt(W1, ph_w1);
  load_wt(W2, ph_w2);
  __syncthreads();

  const int lane = threadIdx.x & 31;
  const int wave = threadIdx.x >> 5;
  const int g = lane >> 4;
  const int lm = lane & 15;
  _Float16* SA = sm + 4 * 128 * KPAD + wave * 16 * KPAD;

  const int wid = blockIdx.x * 8 + wave;
  const int nw = gridDim.x * 8;

  for (int t = wid; t < NTILES_N; t += nw) {
    const int base = t * 16;
    const int node = base + lm;
    h16 Ami[4], Anf[4];
    {
      const float* mrow = mi + (size_t)node * HIDD;
      const float* frow = nf + (size_t)node * FEATD;
#pragma unroll
      for (int c = 0; c < 4; ++c) { Ami[c] = gfragA(mrow, c, g); Anf[c] = gfragA(frow, c, g); }
    }

    // ph0: silu([m_i | h] @ W0 + b0)
#pragma unroll
    for (int nt = 0; nt < 8; ++nt) {
      const int n = nt * 16 + lm;
      const float b = ph_b0[n];
      f8 acc;
#pragma unroll
      for (int j = 0; j < 8; ++j) acc[j] = b;
#pragma unroll
      for (int c = 0; c < 4; ++c) acc = wmma_f16(Ami[c], ldsB(W0a, n, c, g), acc);
#pragma unroll
      for (int c = 0; c < 4; ++c) acc = wmma_f16(Anf[c], ldsB(W0b, n, c, g), acc);
#pragma unroll
      for (int j = 0; j < 8; ++j) SA[(j + 8 * g) * KPAD + n] = (_Float16)silu_f(acc[j]);
    }

    DSFENCE();
    h16 Am[4];
#pragma unroll
    for (int c = 0; c < 4; ++c) Am[c] = ldsA(SA, lm, g, c);
    DSFENCE();

    // ph1
#pragma unroll
    for (int nt = 0; nt < 8; ++nt) {
      const int n = nt * 16 + lm;
      const float b = ph_b1[n];
      f8 acc;
#pragma unroll
      for (int j = 0; j < 8; ++j) acc[j] = b;
#pragma unroll
      for (int c = 0; c < 4; ++c) acc = wmma_f16(Am[c], ldsB(W1, n, c, g), acc);
#pragma unroll
      for (int j = 0; j < 8; ++j) SA[(j + 8 * g) * KPAD + n] = (_Float16)silu_f(acc[j]);
    }

    DSFENCE();
#pragma unroll
    for (int c = 0; c < 4; ++c) Am[c] = ldsA(SA, lm, g, c);
    DSFENCE();

    // ph2 (linear) + residual, store
#pragma unroll
    for (int nt = 0; nt < 8; ++nt) {
      const int n = nt * 16 + lm;
      const float b = ph_b2[n];
      f8 acc;
#pragma unroll
      for (int j = 0; j < 8; ++j) acc[j] = b;
#pragma unroll
      for (int c = 0; c < 4; ++c) acc = wmma_f16(Am[c], ldsB(W2, n, c, g), acc);
#pragma unroll
      for (int j = 0; j < 8; ++j) {
        const int nd = base + j + 8 * g;
        feat_out[(size_t)nd * FEATD + n] = acc[j] + nf[(size_t)nd * FEATD + n];
      }
    }
  }
}

extern "C" void kernel_launch(void* const* d_in, const int* in_sizes, int n_in,
                              void* d_out, int out_size, void* d_ws, size_t ws_size,
                              hipStream_t stream) {
  (void)in_sizes; (void)n_in; (void)out_size; (void)ws_size;
  const float* pos   = (const float*)d_in[0];
  const float* nf    = (const float*)d_in[1];
  const float* pe_w0 = (const float*)d_in[2];
  const float* pe_b0 = (const float*)d_in[3];
  const float* pe_w1 = (const float*)d_in[4];
  const float* pe_b1 = (const float*)d_in[5];
  const float* px_w0 = (const float*)d_in[6];
  const float* px_b0 = (const float*)d_in[7];
  const float* px_w1 = (const float*)d_in[8];
  const float* px_b1 = (const float*)d_in[9];
  const float* px_ow = (const float*)d_in[10];
  const float* px_ob = (const float*)d_in[11];
  const float* e_w   = (const float*)d_in[12];
  const float* e_b   = (const float*)d_in[13];
  const float* ph_w0 = (const float*)d_in[14];
  const float* ph_b0 = (const float*)d_in[15];
  const float* ph_w1 = (const float*)d_in[16];
  const float* ph_b1 = (const float*)d_in[17];
  const float* ph_w2 = (const float*)d_in[18];
  const float* ph_b2 = (const float*)d_in[19];
  const int* senders   = (const int*)d_in[20];
  const int* receivers = (const int*)d_in[21];

  float* out = (float*)d_out;
  float* pos_out = out;                          // [N,3]
  float* feat_out = out + (size_t)NNODES * 3;    // [N,128]
  float* mi = (float*)d_ws;                      // [N,128] scratch accumulator

  egcl_init<<<1024, 256, 0, stream>>>(pos, pos_out, mi);

  const size_t smem_e = (size_t)(5 * 128 * KPAD + 8 * 16 * KPAD) * sizeof(_Float16); // ~204 KB
  egcl_edge<<<640, 256, smem_e, stream>>>(pos, nf, pe_w0, pe_b0, pe_w1, pe_b1,
                                          px_w0, px_b0, px_w1, px_b1, px_ow, px_ob,
                                          e_w, e_b, senders, receivers, mi, pos_out);

  const size_t smem_n = (size_t)(4 * 128 * KPAD + 8 * 16 * KPAD) * sizeof(_Float16); // ~170 KB
  egcl_node<<<160, 256, smem_n, stream>>>(nf, mi, ph_w0, ph_b0, ph_w1, ph_b1,
                                          ph_w2, ph_b2, feat_out);
}
